// Simple_Layer_16947940950219
// MI455X (gfx1250) — matrix operand order repaired and checked by reading, not compiled, not run on hardware
//
#include <hip/hip_runtime.h>

#define DEV __device__ __forceinline__

typedef __attribute__((ext_vector_type(16))) __bf16 v16bf;
typedef __attribute__((ext_vector_type(8)))  float  v8f;
typedef int v4i __attribute__((ext_vector_type(4)));

// Problem constants (match reference)
constexpr int Lseq = 2048;
constexpr int Dmin = 512;
constexpr int Rbuk = 129;   // 2*64+1
constexpr int Bb   = 8;

#if __has_builtin(__builtin_amdgcn_global_load_async_to_lds_b128)
#define HAVE_ASYNC 1
#else
#define HAVE_ASYNC 0
#endif

typedef __attribute__((address_space(1))) v4i* gv4p;   // global v4i*
typedef __attribute__((address_space(3))) v4i* lv4p;   // LDS v4i*

// 16-byte global -> LDS async copy (CDNA5 GLOBAL_LOAD_ASYNC_TO_LDS_B128, ASYNCcnt)
DEV void async_cp16(void* lds, const void* g) {
#if HAVE_ASYNC
    __builtin_amdgcn_global_load_async_to_lds_b128((gv4p)g, (lv4p)lds, 0, 0);
#endif
}
DEV void async_wait0() {
#if HAVE_ASYNC
#if __has_builtin(__builtin_amdgcn_s_wait_asynccnt)
    __builtin_amdgcn_s_wait_asynccnt(0);
#else
    asm volatile("s_wait_asynccnt 0x0" ::: "memory");
#endif
#endif
}

// f32 -> bf16 round-to-nearest-even (scalar)
DEV unsigned short f2bf(float x) {
    unsigned u = __builtin_bit_cast(unsigned, x);
    u += 0x7FFFu + ((u >> 16) & 1u);
    return (unsigned short)(u >> 16);
}
// two f32 -> packed 2x bf16 in one dword via v_perm_b32 (a -> low half)
DEV unsigned packrne2(float a, float b) {
    unsigned ua = __builtin_bit_cast(unsigned, a);
    unsigned ub = __builtin_bit_cast(unsigned, b);
    ua += 0x7FFFu + ((ua >> 16) & 1u);
    ub += 0x7FFFu + ((ub >> 16) & 1u);
    return __builtin_amdgcn_perm(ub, ua, 0x07060302u);
}

// raw bf16 bits from either f32 or bf16 source (scalar)
DEV unsigned short ldraw(const float* p)          { return f2bf(*p); }
DEV unsigned short ldraw(const unsigned short* p) { return *p; }

union FragU { v16bf v; uint4 q[2]; };

// ---- WMMA GEMM tile geometry ----
constexpr int TM = 128, TN = 128, TK = 32;
constexpr int LDA_S = TK + 8;   // ushort stride: 80B rows -> 16B aligned, conflict-padded
constexpr int LDB_S = TK + 8;
constexpr int NIT = (TM * TK) / (256 * 4);   // 4 chunks per thread (4 elems each)

// ---- register staging: A tile / transposed-B tile share geometry [row][k-contig] ----
// Fast path: batch all vector loads first, then convert+store (one wait, pipelined).
template<typename T, bool GK>
DEV void stage_rowmajor(unsigned short* __restrict__ dst, const T* __restrict__ S,
                        int lds, int row_base, int ld, int k0, int K, int tid)
{
    if constexpr (!GK && sizeof(T) == 4) {
        float4 fv[NIT];
        #pragma unroll
        for (int it = 0; it < NIT; ++it) {
            int c = tid + it * 256, r = c >> 3, cc = (c & 7) << 2;
            fv[it] = *(const float4*)&S[(long long)(row_base + r) * ld + (k0 + cc)];
        }
        #pragma unroll
        for (int it = 0; it < NIT; ++it) {
            int c = tid + it * 256, r = c >> 3, cc = (c & 7) << 2;
            uint2 w; w.x = packrne2(fv[it].x, fv[it].y); w.y = packrne2(fv[it].z, fv[it].w);
            *(uint2*)(dst + r * lds + cc) = w;
        }
    } else if constexpr (!GK) {
        uint2 uv[NIT];
        #pragma unroll
        for (int it = 0; it < NIT; ++it) {
            int c = tid + it * 256, r = c >> 3, cc = (c & 7) << 2;
            uv[it] = *(const uint2*)&S[(long long)(row_base + r) * ld + (k0 + cc)];
        }
        #pragma unroll
        for (int it = 0; it < NIT; ++it) {
            int c = tid + it * 256, r = c >> 3, cc = (c & 7) << 2;
            *(uint2*)(dst + r * lds + cc) = uv[it];
        }
    } else {
        #pragma unroll
        for (int it = 0; it < NIT; ++it) {
            int c = tid + it * 256, r = c >> 3, cc = (c & 7) << 2;
            const T* src = &S[(long long)(row_base + r) * ld + (k0 + cc)];
            unsigned short e0 = (k0 + cc + 0 < K) ? ldraw(src + 0) : (unsigned short)0;
            unsigned short e1 = (k0 + cc + 1 < K) ? ldraw(src + 1) : (unsigned short)0;
            unsigned short e2 = (k0 + cc + 2 < K) ? ldraw(src + 2) : (unsigned short)0;
            unsigned short e3 = (k0 + cc + 3 < K) ? ldraw(src + 3) : (unsigned short)0;
            uint2 w;
            w.x = (unsigned)e0 | ((unsigned)e1 << 16);
            w.y = (unsigned)e2 | ((unsigned)e3 << 16);
            *(uint2*)(dst + r * lds + cc) = w;
        }
    }
}

// row-guarded variant (transposed-B with N guard; rows beyond N -> zeros)
template<typename T>
DEV void stage_rowmajor_gn(unsigned short* __restrict__ dst, const T* __restrict__ S,
                           int lds, int row_base, int ld, int k0, int N, int tid)
{
    #pragma unroll
    for (int it = 0; it < NIT; ++it) {
        int c = tid + it * 256, r = c >> 3, cc = (c & 7) << 2;
        int gr = row_base + r;
        uint2 w;
        if (gr >= N) { w.x = 0u; w.y = 0u; }
        else {
            const T* src = &S[(long long)gr * ld + (k0 + cc)];
            if constexpr (sizeof(T) == 4) {
                float4 f = *(const float4*)src;
                w.x = packrne2(f.x, f.y); w.y = packrne2(f.z, f.w);
            } else {
                w = *(const uint2*)src;
            }
        }
        *(uint2*)(dst + r * lds + cc) = w;
    }
}

// async staging: 8 bf16 (16B) per chunk, 2 chunks per thread; no conversion needed
DEV void stage_rowmajor_async(unsigned short* __restrict__ dst, const unsigned short* __restrict__ S,
                              int lds, int row_base, int ld, int k0, int tid)
{
    #pragma unroll
    for (int it = 0; it < (TM * TK) / (256 * 8); ++it) {
        int c  = tid + it * 256;
        int r  = c >> 2;           // 4 chunks per row
        int cc = (c & 3) << 3;     // 0,8,16,24 (ushorts)
        async_cp16(dst + r * lds + cc,
                   S + (long long)(row_base + r) * ld + (k0 + cc));
    }
}

// B in LDS as [n][k]; W row-major [K,N]: batch vector-reads of 4 n's, transpose on store
template<typename T, bool GK>
DEV void stage_Bn(unsigned short* __restrict__ dst, const T* __restrict__ W,
                  int ldw, int col0, int k0, int K, int tid)
{
    if constexpr (!GK && sizeof(T) == 4) {
        float4 fv[NIT];
        #pragma unroll
        for (int it = 0; it < NIT; ++it) {
            int c = tid + it * 256, k = c >> 5, nn = (c & 31) << 2;
            fv[it] = *(const float4*)&W[(long long)(k0 + k) * ldw + (col0 + nn)];
        }
        #pragma unroll
        for (int it = 0; it < NIT; ++it) {
            int c = tid + it * 256, k = c >> 5, nn = (c & 31) << 2;
            dst[(nn + 0) * LDB_S + k] = f2bf(fv[it].x);
            dst[(nn + 1) * LDB_S + k] = f2bf(fv[it].y);
            dst[(nn + 2) * LDB_S + k] = f2bf(fv[it].z);
            dst[(nn + 3) * LDB_S + k] = f2bf(fv[it].w);
        }
    } else if constexpr (!GK) {
        uint2 uv[NIT];
        #pragma unroll
        for (int it = 0; it < NIT; ++it) {
            int c = tid + it * 256, k = c >> 5, nn = (c & 31) << 2;
            uv[it] = *(const uint2*)&W[(long long)(k0 + k) * ldw + (col0 + nn)];
        }
        #pragma unroll
        for (int it = 0; it < NIT; ++it) {
            int c = tid + it * 256, k = c >> 5, nn = (c & 31) << 2;
            dst[(nn + 0) * LDB_S + k] = (unsigned short)uv[it].x;
            dst[(nn + 1) * LDB_S + k] = (unsigned short)(uv[it].x >> 16);
            dst[(nn + 2) * LDB_S + k] = (unsigned short)uv[it].y;
            dst[(nn + 3) * LDB_S + k] = (unsigned short)(uv[it].y >> 16);
        }
    } else {
        #pragma unroll
        for (int it = 0; it < NIT; ++it) {
            int c = tid + it * 256, k = c >> 5, nn = (c & 31) << 2;
            int gk = k0 + k;
            unsigned short e0 = 0, e1 = 0, e2 = 0, e3 = 0;
            if (gk < K) {
                const T* src = &W[(long long)gk * ldw + (col0 + nn)];
                if constexpr (sizeof(T) == 4) {
                    float4 f = *(const float4*)src;
                    e0 = f2bf(f.x); e1 = f2bf(f.y); e2 = f2bf(f.z); e3 = f2bf(f.w);
                } else {
                    uint2 u = *(const uint2*)src;
                    e0 = (unsigned short)u.x; e1 = (unsigned short)(u.x >> 16);
                    e2 = (unsigned short)u.y; e3 = (unsigned short)(u.y >> 16);
                }
            }
            dst[(nn + 0) * LDB_S + k] = e0;
            dst[(nn + 1) * LDB_S + k] = e1;
            dst[(nn + 2) * LDB_S + k] = e2;
            dst[(nn + 3) * LDB_S + k] = e3;
        }
    }
}

// OUT_MODE: 0 = store bf16, 1 = store f32, 2 = accumulate f32,
//           3 = scores epilogue (add rel-pos bias, scale by 1/sqrt(D))
template<typename AT, typename WTy, bool W_TRANS, int OUT_MODE, bool GN, bool GK>
__global__ __launch_bounds__(256)
void gemm_wmma(const AT* __restrict__ Ag, int lda, long long aBS,
               const WTy* __restrict__ Wg, int ldw, long long wBS,
               void* __restrict__ Cg, int ldc, long long cBS,
               int M, int N, int K,
               const float* __restrict__ relS, long long relBS)
{
    (void)M;
    __shared__ __align__(16) unsigned short Al[2][TM * LDA_S];
    __shared__ __align__(16) unsigned short Bl[2][TN * LDB_S];

    constexpr bool ASY_A = (HAVE_ASYNC != 0) && (sizeof(AT) == 2) && !GK;
    constexpr bool ASY_B = (HAVE_ASYNC != 0) && W_TRANS && (sizeof(WTy) == 2) && !GN && !GK;

    const int tid  = threadIdx.x;
    const int lane = tid & 31;
    const int wv   = tid >> 5;
    const int wm   = wv >> 1;        // 0..3 : wave row (32 rows each)
    const int wn   = wv & 1;         // 0..1 : wave col (64 cols each)
    const int b    = blockIdx.z;
    const int row0 = blockIdx.y * TM;
    const int col0 = blockIdx.x * TN;

    const AT*  A = Ag + (long long)b * aBS;
    const WTy* W = Wg + (long long)b * wBS;

    v8f acc[2][4];
    #pragma unroll
    for (int mi = 0; mi < 2; ++mi)
        #pragma unroll
        for (int ni = 0; ni < 4; ++ni)
            #pragma unroll
            for (int e = 0; e < 8; ++e) acc[mi][ni][e] = 0.0f;

    // prologue: stage first K-tile into buffer 0
    if constexpr (ASY_A) stage_rowmajor_async(Al[0], (const unsigned short*)A, LDA_S, row0, lda, 0, tid);
    else                 stage_rowmajor<AT, GK>(Al[0], A, LDA_S, row0, lda, 0, K, tid);
    if constexpr (W_TRANS) {
        if constexpr (ASY_B)      stage_rowmajor_async(Bl[0], (const unsigned short*)W, LDB_S, col0, ldw, 0, tid);
        else if constexpr (GN)    stage_rowmajor_gn<WTy>(Bl[0], W, LDB_S, col0, ldw, 0, N, tid);
        else                      stage_rowmajor<WTy, GK>(Bl[0], W, LDB_S, col0, ldw, 0, K, tid);
    } else {
        stage_Bn<WTy, GK>(Bl[0], W, ldw, col0, 0, K, tid);
    }
    if constexpr (ASY_A || ASY_B) async_wait0();
    __syncthreads();

    int cur = 0;
    for (int k0 = 0; k0 < K; k0 += TK) {
        // ---- load fragments per documented CDNA5 16-bit layouts ----
        // A 16x32: lanes<16 -> k {0..7,16..23}; lanes>=16 -> k {8..15,24..31}
        FragU af[2];
        #pragma unroll
        for (int mi = 0; mi < 2; ++mi) {
            int m  = wm * 32 + mi * 16 + (lane & 15);
            int kb = (lane >> 4) << 3;                 // 0 or 8
            const unsigned short* p = &Al[cur][m * LDA_S + kb];
            af[mi].q[0] = *(const uint4*)(p);
            af[mi].q[1] = *(const uint4*)(p + 16);
        }
        // B 32x16: lanes<16 -> k {0..7,16..23}; lanes>=16 -> k {8..15,24..31} at column n
        FragU bfg[4];
        #pragma unroll
        for (int ni = 0; ni < 4; ++ni) {
            int n  = wn * 64 + ni * 16 + (lane & 15);
            int kb = (lane >> 4) << 3;                 // 0 or 8
            const unsigned short* p = &Bl[cur][n * LDB_S + kb];
            bfg[ni].q[0] = *(const uint4*)(p);
            bfg[ni].q[1] = *(const uint4*)(p + 16);
        }

        #pragma unroll
        for (int mi = 0; mi < 2; ++mi)
            #pragma unroll
            for (int ni = 0; ni < 4; ++ni)
                acc[mi][ni] = __builtin_amdgcn_wmma_f32_16x16x32_bf16(
                    false, af[mi].v, false, bfg[ni].v,
                    (short)0, acc[mi][ni], false, false);

        // ---- stage next K-tile into the other buffer (overlaps with WMMA) ----
        int k1 = k0 + TK;
        if (k1 < K) {
            if constexpr (ASY_A) stage_rowmajor_async(Al[cur ^ 1], (const unsigned short*)A, LDA_S, row0, lda, k1, tid);
            else                 stage_rowmajor<AT, GK>(Al[cur ^ 1], A, LDA_S, row0, lda, k1, K, tid);
            if constexpr (W_TRANS) {
                if constexpr (ASY_B)      stage_rowmajor_async(Bl[cur ^ 1], (const unsigned short*)W, LDB_S, col0, ldw, k1, tid);
                else if constexpr (GN)    stage_rowmajor_gn<WTy>(Bl[cur ^ 1], W, LDB_S, col0, ldw, k1, N, tid);
                else                      stage_rowmajor<WTy, GK>(Bl[cur ^ 1], W, LDB_S, col0, ldw, k1, K, tid);
            } else {
                stage_Bn<WTy, GK>(Bl[cur ^ 1], W, ldw, col0, k1, K, tid);
            }
        }
        if constexpr (ASY_A || ASY_B) async_wait0();
        __syncthreads();
        cur ^= 1;
    }

    // ---- epilogue: C 16x16 f32 layout: VGPR v -> M=v (lanes<16) / v+8 ----
    #pragma unroll
    for (int mi = 0; mi < 2; ++mi) {
        int mb = row0 + wm * 32 + mi * 16 + ((lane >= 16) ? 8 : 0);
        #pragma unroll
        for (int ni = 0; ni < 4; ++ni) {
            int n = col0 + wn * 64 + ni * 16 + (lane & 15);
            if (!GN || n < N) {
                #pragma unroll
                for (int v = 0; v < 8; ++v) {
                    int m = mb + v;
                    float val = acc[mi][ni][v];
                    long long off = (long long)b * cBS + (long long)m * ldc + n;
                    if constexpr (OUT_MODE == 0) {
                        ((unsigned short*)Cg)[off] = f2bf(val);
                    } else if constexpr (OUT_MODE == 1) {
                        ((float*)Cg)[off] = val;
                    } else if constexpr (OUT_MODE == 2) {
                        float* p = &((float*)Cg)[off];
                        *p = *p + val;
                    } else {
                        int d = n - m;                 // j - i
                        d = d < -64 ? -64 : (d > 64 ? 64 : d);
                        float bias = relS[(long long)b * relBS
                                          + (long long)m * Rbuk + (d + 64)];
                        ((float*)Cg)[off] = (val + bias) * 0.04419417382415922f;
                    }
                }
            }
        }
    }
}

// ---- per-row softmax + relative-bucket scatter ----
__global__ __launch_bounds__(256)
void softmax_rel(float* __restrict__ attn, float* __restrict__ relA)
{
    __shared__ float red[256];
    __shared__ float bins[Rbuk];
    const int row = blockIdx.x;            // 0 .. B*L-1
    const int i   = row & (Lseq - 1);
    const int tid = threadIdx.x;
    float* S = attn + (long long)row * Lseq;

    for (int r = tid; r < Rbuk; r += 256) bins[r] = 0.0f;

    float v[8];
    float mx = -3.4e38f;
    #pragma unroll
    for (int k = 0; k < 8; ++k) { v[k] = S[k * 256 + tid]; mx = fmaxf(mx, v[k]); }
    red[tid] = mx; __syncthreads();
    for (int s = 128; s > 0; s >>= 1) {
        if (tid < s) red[tid] = fmaxf(red[tid], red[tid + s]);
        __syncthreads();
    }
    mx = red[0]; __syncthreads();

    float sum = 0.0f;
    #pragma unroll
    for (int k = 0; k < 8; ++k) { v[k] = __expf(v[k] - mx); sum += v[k]; }
    red[tid] = sum; __syncthreads();
    for (int s = 128; s > 0; s >>= 1) {
        if (tid < s) red[tid] += red[tid + s];
        __syncthreads();
    }
    const float inv = 1.0f / red[0];

    #pragma unroll
    for (int k = 0; k < 8; ++k) {
        int j = k * 256 + tid;
        float p = v[k] * inv;
        S[j] = p;
        int d = j - i; d = d < -64 ? -64 : (d > 64 ? 64 : d);
        atomicAdd(&bins[d + 64], p);      // ds_add_f32
    }
    __syncthreads();
    for (int r = tid; r < Rbuk; r += 256)
        relA[(long long)row * Rbuk + r] = bins[r];
}

// ---- per-row LayerNorm over D=512 ----
__global__ __launch_bounds__(256)
void ln_kernel(const float* __restrict__ ctx, const float* __restrict__ gamma,
               const float* __restrict__ beta, float* __restrict__ Y)
{
    __shared__ float r1[256], r2[256];
    const long long row = blockIdx.x;
    const int tid = threadIdx.x;
    const float* x = ctx + row * Dmin;
    float a = x[tid], c = x[tid + 256];
    r1[tid] = a + c; r2[tid] = a * a + c * c;
    __syncthreads();
    for (int s = 128; s > 0; s >>= 1) {
        if (tid < s) { r1[tid] += r1[tid + s]; r2[tid] += r2[tid + s]; }
        __syncthreads();
    }
    float mu  = r1[0] * (1.0f / 512.0f);
    float var = r2[0] * (1.0f / 512.0f) - mu * mu;
    float rs  = rsqrtf(var + 1e-5f);
    float* y = Y + row * Dmin;
    y[tid]       = (a - mu) * rs * gamma[tid]       + beta[tid];
    y[tid + 256] = (c - mu) * rs * gamma[tid + 256] + beta[tid + 256];
}

extern "C" void kernel_launch(void* const* d_in, const int* in_sizes, int n_in,
                              void* d_out, int out_size, void* d_ws, size_t ws_size,
                              hipStream_t stream)
{
    (void)in_sizes; (void)n_in; (void)out_size; (void)ws_size;
    const float* Xmaj = (const float*)d_in[0];   // [B,L,1024]
    const float* Xmin = (const float*)d_in[1];   // [B,L,512]
    const float* Wq   = (const float*)d_in[2];   // [1024,512]
    const float* Wk   = (const float*)d_in[3];   // [512,512]
    const float* Wv   = (const float*)d_in[4];   // [512,512]
    const float* aK   = (const float*)d_in[5];   // [129,512]
    const float* aV   = (const float*)d_in[6];   // [129,512]
    const float* g    = (const float*)d_in[7];   // [512]
    const float* be   = (const float*)d_in[8];   // [512]

    const long long Mtot = (long long)Bb * Lseq;           // 16384
    const long long bsQK = (long long)Lseq * Dmin;         // per-batch q/k/v stride
    const long long bsAA = (long long)Lseq * Lseq;         // per-batch attn stride

    // workspace layout
    unsigned short* qb = (unsigned short*)d_ws;            // bf16 [16384,512]
    unsigned short* kb = qb + Mtot * Dmin;
    unsigned short* vb = kb + Mtot * Dmin;
    float* relS = (float*)(vb + Mtot * Dmin);              // [16384,129]
    float* relA = relS + Mtot * Rbuk;                      // [16384,129]
    float* ctx  = relA + Mtot * Rbuk;                      // [16384,512]

    float* Y    = (float*)d_out;                           // [16384,512]
    float* attn = Y + Mtot * Dmin;                         // [8,2048,2048]

    dim3 blk(256);

    // 1-3) projections -> bf16 (fully unguarded fast path)
    gemm_wmma<float, float, false, 0, false, false><<<dim3(Dmin / TN, Mtot / TM, 1), blk, 0, stream>>>(
        Xmaj, 1024, 0, Wq, 512, 0, qb, 512, 0, (int)Mtot, Dmin, 1024, nullptr, 0);
    gemm_wmma<float, float, false, 0, false, false><<<dim3(Dmin / TN, Mtot / TM, 1), blk, 0, stream>>>(
        Xmin, 512, 0, Wk, 512, 0, kb, 512, 0, (int)Mtot, Dmin, 512, nullptr, 0);
    gemm_wmma<float, float, false, 0, false, false><<<dim3(Dmin / TN, Mtot / TM, 1), blk, 0, stream>>>(
        Xmin, 512, 0, Wv, 512, 0, vb, 512, 0, (int)Mtot, Dmin, 512, nullptr, 0);

    // 4) rel_scores[b,i,r] = q . aK[r]  (aK read transposed; N=129 guarded; A async bf16)
    gemm_wmma<unsigned short, float, true, 1, true, false><<<dim3((Rbuk + TN - 1) / TN, Mtot / TM, 1), blk, 0, stream>>>(
        qb, 512, 0, aK, 512, 0, relS, Rbuk, 0, (int)Mtot, Rbuk, Dmin, nullptr, 0);

    // 5) scores = (q k^T + rel bias) / sqrt(D)  (batched; A and B via async-to-LDS)
    gemm_wmma<unsigned short, unsigned short, true, 3, false, false><<<dim3(Lseq / TN, Lseq / TM, Bb), blk, 0, stream>>>(
        qb, 512, bsQK, kb, 512, bsQK, attn, Lseq, bsAA,
        Lseq, Lseq, Dmin, relS, (long long)Lseq * Rbuk);

    // 6) row softmax (in place) + rel_attn bucket scatter
    softmax_rel<<<dim3((unsigned)Mtot), blk, 0, stream>>>(attn, relA);

    // 7) ctx = attn @ v (batched; attn f32 -> bf16 on stage, v already bf16)
    gemm_wmma<float, unsigned short, false, 1, false, false><<<dim3(Dmin / TN, Lseq / TM, Bb), blk, 0, stream>>>(
        attn, Lseq, bsAA, vb, 512, bsQK, ctx, 512, bsQK,
        Lseq, Dmin, Lseq, nullptr, 0);

    // 8) ctx += rel_attn @ aV  (K=129 guarded)
    gemm_wmma<float, float, false, 2, false, true><<<dim3(Dmin / TN, Mtot / TM, 1), blk, 0, stream>>>(
        relA, Rbuk, 0, aV, 512, 0, ctx, 512, 0, (int)Mtot, Dmin, Rbuk, nullptr, 0);

    // 9) LayerNorm -> Y
    ln_kernel<<<dim3((unsigned)Mtot), blk, 0, stream>>>(ctx, g, be, Y);
}
